// GNNEncoder_55413668053238
// MI455X (gfx1250) — matrix operand order repaired and checked by reading, not compiled, not run on hardware
//
#include <hip/hip_runtime.h>
#include <hip/hip_bf16.h>
#include <math.h>

// ---------------- WMMA types (CDNA5 / gfx1250, wave32) ----------------
typedef __attribute__((ext_vector_type(16))) __bf16 v16bf;
typedef __attribute__((ext_vector_type(8)))  float  v8f;

// ---------------- constants ----------------
#define GAT_N 20000
#define GAT_E 640000
#define GAT_ETOT (GAT_E + GAT_N)
#define GAT_B 32
#define GAT_H 4

// =====================================================================
// Async global->LDS copy helpers (CDNA5 async data path).
// Low 32 bits of a generic pointer to __shared__ are the LDS byte address
// (ISA 10.2).
// =====================================================================
static __device__ inline unsigned lds_lo(const void* p) {
  return (unsigned)(size_t)p;
}
static __device__ inline void async_b128(unsigned lds, const float* g) {
  asm volatile("global_load_async_to_lds_b128 %0, %1, off"
               :: "v"(lds), "v"(g) : "memory");
}
static __device__ inline void async_b32(unsigned lds, const float* g) {
  asm volatile("global_load_async_to_lds_b32 %0, %1, off"
               :: "v"(lds), "v"(g) : "memory");
}
static __device__ inline void wait_async0() {
  asm volatile("s_wait_asynccnt 0" ::: "memory");
}

// Load 16 contiguous f32 from LDS (4x ds_load_b128) and convert to bf16 frag
static __device__ inline v16bf cvt_frag(const float* p) {
  const float4* q = (const float4*)p;
  float4 a = q[0], b = q[1], c = q[2], d = q[3];
  v16bf f;
  f[0]  = (__bf16)a.x; f[1]  = (__bf16)a.y; f[2]  = (__bf16)a.z; f[3]  = (__bf16)a.w;
  f[4]  = (__bf16)b.x; f[5]  = (__bf16)b.y; f[6]  = (__bf16)b.z; f[7]  = (__bf16)b.w;
  f[8]  = (__bf16)c.x; f[9]  = (__bf16)c.y; f[10] = (__bf16)c.z; f[11] = (__bf16)c.w;
  f[12] = (__bf16)d.x; f[13] = (__bf16)d.y; f[14] = (__bf16)d.z; f[15] = (__bf16)d.w;
  return f;
}

// =====================================================================
// GEMM: C[M,Ncol] = A[M,K] @ W[K,Ncol], f32 in/out, bf16 WMMA compute.
// 256 threads = 8 waves arranged 4(row)x2(col); block tile 128x64, K-step 32.
// Each wave: 2 A-frags x 2 B-frags -> 4 WMMAs per K-step.
// LDS tiles are stored in *fragment order* (A: middle two 8-wide K-blocks
// swapped; B: transposed [n][k], same K swap) so each lane's 16 elements are contiguous.
// Double-buffered: async DMA of tile k+1 overlaps WMMA of tile k; a single
// barrier per K-step both publishes the waited tile and retires the buffer
// being refilled (every wave does s_wait_asynccnt 0 before signaling).
// =====================================================================
#define TM 128
#define TN 64
#define TKF 32
#define PADK 36   // f32 row stride (144B: 16B-aligned, bank-skewed)

__global__ __launch_bounds__(256)
void gemm_bf16_wmma(const float* __restrict__ A, const float* __restrict__ W,
                    float* __restrict__ C, int M, int K, int Ncol) {
  __shared__ float As32[2][TM][PADK];   // [buf][row][perm K]   2 x 18.4 KB
  __shared__ float Bs32[2][TN][PADK];   // [buf][col n][perm K]      2 x  9.2 KB

  const int tid  = threadIdx.x;
  const int wave = tid >> 5;
  const int lane = tid & 31;
  const int hf   = lane >> 4;           // half-wave
  const int mrow = lane & 15;
  const int wr   = wave & 3;            // wave row (0..3) -> 32 rows each
  const int wc   = wave >> 2;           // wave col (0..1) -> 32 cols each

  const int rowBase = blockIdx.x * TM;
  const int colBase = blockIdx.y * TN;

  // ---- per-thread staging descriptors (hoisted out of the K loop) ----
  const float* gpa[4]; unsigned la[4];
#pragma unroll
  for (int u = 0; u < 4; ++u) {
    int j = tid + u * 256;                    // over TM*8 4-float chunks
    int r = j >> 3, c4 = j & 7;
    int blk  = c4 >> 1;                       // LDS position block
    int kblk = (blk == 1) ? 2 : ((blk == 2) ? 1 : blk);   // perm 0,2,1,3
    int gr = rowBase + r; if (gr >= M) gr = 0;            // clamp (unused rows)
    gpa[u] = A + (size_t)gr * K + kblk * 8 + (c4 & 1) * 4;
    la[u]  = lds_lo(&As32[0][r][c4 * 4]);
  }
  const float* gpb[8]; unsigned lb[8];
#pragma unroll
  for (int u = 0; u < 8; ++u) {
    int j = tid + u * 256;                    // over TKF*TN elements
    int n = j & 63, k = j >> 6;               // consecutive tid -> n (coalesced)
    gpb[u] = W + (size_t)k * Ncol + colBase + n;
    lb[u]  = lds_lo(&Bs32[0][n][(k & 7) | ((k & 8) << 1) | ((k & 16) >> 1)]);
  }
  const unsigned ASZ = TM * PADK * 4;
  const unsigned BSZ = TN * PADK * 4;

  auto issue = [&](int buf, int step) {
    const int ka = step * TKF;
#pragma unroll
    for (int u = 0; u < 4; ++u) async_b128(la[u] + buf * ASZ, gpa[u] + ka);
    const size_t kb = (size_t)ka * Ncol;
#pragma unroll
    for (int u = 0; u < 8; ++u) async_b32(lb[u] + buf * BSZ, gpb[u] + kb);
  };

  v8f acc00 = {}, acc01 = {}, acc10 = {}, acc11 = {};

  const int steps = K / TKF;
  issue(0, 0);
  for (int s = 0; s < steps; ++s) {
    const int cur = s & 1;
    wait_async0();            // own async writes to LDS complete
    __syncthreads();          // tile `cur` published; buffer 1-cur retired
    if (s + 1 < steps) issue(1 - cur, s + 1);   // overlap fetch with WMMA

    v16bf a0 = cvt_frag(&As32[cur][wr * 32 + mrow][hf * 16]);
    v16bf a1 = cvt_frag(&As32[cur][wr * 32 + 16 + mrow][hf * 16]);
    v16bf b0 = cvt_frag(&Bs32[cur][wc * 32 + mrow][hf * 16]);
    v16bf b1 = cvt_frag(&Bs32[cur][wc * 32 + 16 + mrow][hf * 16]);

    acc00 = __builtin_amdgcn_wmma_f32_16x16x32_bf16(false, a0, false, b0, (short)0, acc00, false, false);
    acc01 = __builtin_amdgcn_wmma_f32_16x16x32_bf16(false, a0, false, b1, (short)0, acc01, false, false);
    acc10 = __builtin_amdgcn_wmma_f32_16x16x32_bf16(false, a1, false, b0, (short)0, acc10, false, false);
    acc11 = __builtin_amdgcn_wmma_f32_16x16x32_bf16(false, a1, false, b1, (short)0, acc11, false, false);
  }

  // ---- epilogue: elem r -> M = r + 8*hf, N = lane%16 ----
  const int  tileRow = rowBase + wr * 32 + 8 * hf;
  const int  colOut  = colBase + wc * 32 + mrow;
  const bool full    = (rowBase + TM) <= M;
  float* cp0 = C + (size_t)tileRow * Ncol + colOut;
#pragma unroll
  for (int ri = 0; ri < 2; ++ri) {
    float* cp = cp0 + (size_t)(ri * 16) * Ncol;
#pragma unroll
    for (int r = 0; r < 8; ++r) {
      if (full || (tileRow + ri * 16 + r) < M) {
        cp[0]  = ri ? acc10[r] : acc00[r];
        cp[16] = ri ? acc11[r] : acc01[r];
      }
      cp += Ncol;
    }
  }
}

// =====================================================================
// Small elementwise / graph kernels
// =====================================================================
__global__ void fill_f32(float* __restrict__ p, float v, long long n) {
  long long i = (long long)blockIdx.x * blockDim.x + threadIdx.x;
  if (i < n) p[i] = v;
}

__global__ void build_edges(const long long* __restrict__ ei,
                            int* __restrict__ srcb, int* __restrict__ dstb) {
  int i = blockIdx.x * blockDim.x + threadIdx.x;
  if (i >= GAT_ETOT) return;
  if (i < GAT_E) { srcb[i] = (int)ei[i]; dstb[i] = (int)ei[GAT_E + i]; }
  else           { srcb[i] = i - GAT_E;  dstb[i] = i - GAT_E; }
}

// alpha_{s,d}[n,h] = sum_c h[n,h,c] * a_{s,d}[h,c]   (float4 vectorized)
__global__ void alpha_kernel(const float* __restrict__ h,
                             const float* __restrict__ a_src,
                             const float* __restrict__ a_dst,
                             float* __restrict__ as_, float* __restrict__ ad_,
                             int C) {
  int i = blockIdx.x * blockDim.x + threadIdx.x;        // over N*H
  if (i >= GAT_N * GAT_H) return;
  int n = i >> 2, hh = i & 3;
  const float4* hp  = (const float4*)(h + ((size_t)n * GAT_H + hh) * C);
  const float4* asr = (const float4*)(a_src + hh * C);
  const float4* adr = (const float4*)(a_dst + hh * C);
  float s = 0.f, d = 0.f;
  const int C4 = C >> 2;
  for (int c = 0; c < C4; ++c) {
    float4 v = hp[c], u = asr[c], w = adr[c];
    s += v.x * u.x + v.y * u.y + v.z * u.z + v.w * u.w;
    d += v.x * w.x + v.y * w.y + v.z * w.z + v.w * w.w;
  }
  as_[i] = s; ad_[i] = d;
}

// float atomic max via monotone int/uint trick (exact, no CAS loop)
__device__ inline void atomicMaxF(float* addr, float v) {
  if (v >= 0.0f) atomicMax((int*)addr, __float_as_int(v));
  else           atomicMin((unsigned int*)addr, (unsigned int)__float_as_int(v));
}

__global__ void edge_logits(const float* __restrict__ as_, const float* __restrict__ ad_,
                            const int* __restrict__ src, const int* __restrict__ dst,
                            float* __restrict__ ebuf, float* __restrict__ mmax) {
  int i = blockIdx.x * blockDim.x + threadIdx.x;        // over Etot*H
  if (i >= GAT_ETOT * GAT_H) return;
  int ed = i >> 2, hh = i & 3;
  float v = as_[src[ed] * GAT_H + hh] + ad_[dst[ed] * GAT_H + hh];
  v = (v > 0.f) ? v : 0.2f * v;                         // leaky_relu(0.2)
  ebuf[i] = v;
  atomicMaxF(&mmax[dst[ed] * GAT_H + hh], v);
}

__global__ void edge_exp(float* __restrict__ ebuf, const float* __restrict__ mmax,
                         const int* __restrict__ dst, float* __restrict__ denom) {
  int i = blockIdx.x * blockDim.x + threadIdx.x;        // over Etot*H
  if (i >= GAT_ETOT * GAT_H) return;
  int ed = i >> 2, hh = i & 3;
  float p = __expf(ebuf[i] - mmax[dst[ed] * GAT_H + hh]);
  ebuf[i] = p;
  atomicAdd(&denom[dst[ed] * GAT_H + hh], p);
}

// out[dst] += alpha * h[src]; float4 gathers, f32 atomics (resolved in L2)
__global__ void edge_scatter(const float* __restrict__ h, const float* __restrict__ ebuf,
                             const float* __restrict__ denom,
                             const int* __restrict__ src, const int* __restrict__ dst,
                             float* __restrict__ accum, int C) {
  const int HC  = GAT_H * C;
  const int HC4 = HC >> 2;
  long long gi = (long long)blockIdx.x * blockDim.x + threadIdx.x;
  if (gi >= (long long)GAT_ETOT * HC4) return;
  int ed = (int)(gi / HC4);
  int f4 = (int)(gi % HC4);
  int hh = (f4 << 2) / C;
  int s = src[ed], d = dst[ed];
  float alpha = ebuf[ed * GAT_H + hh] / (denom[d * GAT_H + hh] + 1e-16f);
  const float* hrow = h + (size_t)s * HC;
  __builtin_prefetch(hrow + f4 * 4, 0, 3);              // global_prefetch_b8
  float4 hv = ((const float4*)hrow)[f4];
  float* ap = accum + (size_t)d * HC + f4 * 4;
  atomicAdd(ap + 0, hv.x * alpha);
  atomicAdd(ap + 1, hv.y * alpha);
  atomicAdd(ap + 2, hv.z * alpha);
  atomicAdd(ap + 3, hv.w * alpha);
}

__global__ void bias_elu(const float* __restrict__ accum, const float* __restrict__ b,
                         float* __restrict__ out, int HC) {
  long long i = (long long)blockIdx.x * blockDim.x + threadIdx.x;
  if (i >= (long long)GAT_N * HC) return;
  float v = accum[i] + b[(int)(i % HC)];
  out[i] = (v > 0.f) ? v : (__expf(v) - 1.0f);
}

// =====================================================================
// Pooling + FC
// =====================================================================
__global__ void pool_cnt(const long long* __restrict__ batch, float* __restrict__ cnt) {
  int n = blockIdx.x * blockDim.x + threadIdx.x;
  if (n < GAT_N) atomicAdd(&cnt[(int)batch[n]], 1.0f);
}

__global__ void pool_accum(const float* __restrict__ h, const long long* __restrict__ batch,
                           float* __restrict__ pmean, float* __restrict__ pmax) {
  long long i = (long long)blockIdx.x * blockDim.x + threadIdx.x;
  if (i >= (long long)GAT_N * 512) return;
  int n = (int)(i >> 9), f = (int)(i & 511);
  int b = (int)batch[n];
  float v = h[i];
  atomicAdd(&pmean[b * 512 + f], v);
  atomicMaxF(&pmax[b * 512 + f], v);
}

__global__ void pool_final(const float* __restrict__ pmean, const float* __restrict__ pmax,
                           const float* __restrict__ cnt, float* __restrict__ g) {
  int i = blockIdx.x * blockDim.x + threadIdx.x;        // over B*512
  if (i >= GAT_B * 512) return;
  int b = i >> 9, f = i & 511;
  float c  = fmaxf(cnt[b], 1.0f);
  float mn = pmean[i] / c;
  float mx = pmax[i];
  if (!isfinite(mx)) mx = 0.0f;
  g[b * 1024 + f]       = mn;
  g[b * 1024 + 512 + f] = mx;
}

__global__ void fc_kernel(const float* __restrict__ g, const float* __restrict__ fcW,
                          const float* __restrict__ fcb, float* __restrict__ out) {
  int i = blockIdx.x * blockDim.x + threadIdx.x;        // over B*512
  if (i >= GAT_B * 512) return;
  int b = i >> 9, o = i & 511;
  const float* gr = g + b * 1024;
  float s = fcb[o];
  for (int k = 0; k < 1024; ++k) s += gr[k] * fcW[k * 512 + o];
  out[i] = s;
}

// =====================================================================
// Host-side orchestration
// =====================================================================
static inline unsigned nblk(long long n, int t) { return (unsigned)((n + t - 1) / t); }

extern "C" void kernel_launch(void* const* d_in, const int* in_sizes, int n_in,
                              void* d_out, int out_size, void* d_ws, size_t ws_size,
                              hipStream_t stream) {
  (void)in_sizes; (void)n_in; (void)out_size; (void)ws_size;
  const float*     x     = (const float*)d_in[0];
  const long long* ei    = (const long long*)d_in[1];
  const long long* batch = (const long long*)d_in[2];
  const float* W[4]  = {(const float*)d_in[3],  (const float*)d_in[7],
                        (const float*)d_in[11], (const float*)d_in[15]};
  const float* As[4] = {(const float*)d_in[4],  (const float*)d_in[8],
                        (const float*)d_in[12], (const float*)d_in[16]};
  const float* Ad[4] = {(const float*)d_in[5],  (const float*)d_in[9],
                        (const float*)d_in[13], (const float*)d_in[17]};
  const float* Bv[4] = {(const float*)d_in[6],  (const float*)d_in[10],
                        (const float*)d_in[14], (const float*)d_in[18]};
  const float* fcW = (const float*)d_in[19];
  const float* fcb = (const float*)d_in[20];
  float* out = (float*)d_out;

  // -------- workspace layout --------
  float* xin    = (float*)d_ws;                       // N*512 (layer inputs)
  float* hbuf   = xin    + (size_t)GAT_N * 512;       // N*512 (h = x@W)
  float* accum  = hbuf   + (size_t)GAT_N * 512;       // N*512 (message accum)
  float* alph_s = accum  + (size_t)GAT_N * 512;       // N*H
  float* alph_d = alph_s + (size_t)GAT_N * GAT_H;
  float* mmax   = alph_d + (size_t)GAT_N * GAT_H;
  float* denom  = mmax   + (size_t)GAT_N * GAT_H;
  float* ebuf   = denom  + (size_t)GAT_N * GAT_H;     // Etot*H
  float* pmean  = ebuf   + (size_t)GAT_ETOT * GAT_H;  // B*512
  float* pmax   = pmean  + GAT_B * 512;
  float* pcnt   = pmax   + GAT_B * 512;               // B
  float* gbuf   = pcnt   + GAT_B;                     // B*1024
  int*   srcb   = (int*)(gbuf + GAT_B * 1024);        // Etot
  int*   dstb   = srcb + GAT_ETOT;                    // Etot

  build_edges<<<nblk(GAT_ETOT, 256), 256, 0, stream>>>(ei, srcb, dstb);

  const int Kdim[4] = {128, 256, 256, 256};
  const int Cdim[4] = {64, 64, 64, 128};
  const float* Ain = x;

  for (int l = 0; l < 4; ++l) {
    const int C  = Cdim[l];
    const int HC = GAT_H * C;

    dim3 gg((GAT_N + TM - 1) / TM, HC / TN);
    gemm_bf16_wmma<<<gg, 256, 0, stream>>>(Ain, W[l], hbuf, GAT_N, Kdim[l], HC);

    alpha_kernel<<<nblk((long long)GAT_N * GAT_H, 256), 256, 0, stream>>>(
        hbuf, As[l], Ad[l], alph_s, alph_d, C);

    fill_f32<<<nblk((long long)GAT_N * GAT_H, 256), 256, 0, stream>>>(mmax, -INFINITY, (long long)GAT_N * GAT_H);
    fill_f32<<<nblk((long long)GAT_N * GAT_H, 256), 256, 0, stream>>>(denom, 0.0f, (long long)GAT_N * GAT_H);
    fill_f32<<<nblk((long long)GAT_N * HC, 256), 256, 0, stream>>>(accum, 0.0f, (long long)GAT_N * HC);

    edge_logits<<<nblk((long long)GAT_ETOT * GAT_H, 256), 256, 0, stream>>>(
        alph_s, alph_d, srcb, dstb, ebuf, mmax);
    edge_exp<<<nblk((long long)GAT_ETOT * GAT_H, 256), 256, 0, stream>>>(
        ebuf, mmax, dstb, denom);
    edge_scatter<<<nblk((long long)GAT_ETOT * (HC >> 2), 256), 256, 0, stream>>>(
        hbuf, ebuf, denom, srcb, dstb, accum, C);

    bias_elu<<<nblk((long long)GAT_N * HC, 256), 256, 0, stream>>>(accum, Bv[l], xin, HC);
    Ain = xin;
  }

  // -------- pooling (mean + max per batch) --------
  fill_f32<<<nblk(GAT_B * 512, 256), 256, 0, stream>>>(pmean, 0.0f, GAT_B * 512);
  fill_f32<<<nblk(GAT_B * 512, 256), 256, 0, stream>>>(pmax, -INFINITY, GAT_B * 512);
  fill_f32<<<1, 32, 0, stream>>>(pcnt, 0.0f, GAT_B);
  pool_cnt<<<nblk(GAT_N, 256), 256, 0, stream>>>(batch, pcnt);
  pool_accum<<<nblk((long long)GAT_N * 512, 256), 256, 0, stream>>>(xin, batch, pmean, pmax);
  pool_final<<<nblk(GAT_B * 512, 256), 256, 0, stream>>>(pmean, pmax, pcnt, gbuf);

  // -------- final FC: [32,1024] @ [1024,512] + b --------
  fc_kernel<<<nblk(GAT_B * 512, 256), 256, 0, stream>>>(gbuf, fcW, fcb, out);
}
